// ColumnParallelLinear_46909632807570
// MI455X (gfx1250) — matrix operand order repaired and checked by reading, not compiled, not run on hardware
//
#include <hip/hip_runtime.h>

// ColumnParallelLinear forward: out[s,b,o] = sum_h in[s,b,h] * w[o,h]
// M = S*B = 8192, N = OUT = 4096, K = H = 1024, all fp32.
//
// Strategy (MI455X / gfx1250): bf16x3 split-precision GEMM on the WMMA units.
//   x = hi(x) + lo(x), hi = bf16(x), lo = bf16(x - hi)
//   A@B ~= Ah@Bh + Ah@Bl + Al@Bh   (lo*lo term ~2^-16 relative, dropped)
// Uses v_wmma_f32_16x16x32_bf16 (wave32), LDS-staged tiles, fp32 accumulate.

typedef __bf16 bf16_t;
typedef __attribute__((ext_vector_type(16))) __bf16 v16bf;
typedef __attribute__((ext_vector_type(8)))  __bf16 v8bf;
typedef __attribute__((ext_vector_type(4)))  __bf16 v4bf;
typedef __attribute__((ext_vector_type(8)))  float  v8f;

namespace {
constexpr int S_   = 2048;
constexpr int B_   = 4;
constexpr int H_   = 1024;
constexpr int OUT_ = 4096;

constexpr int Mtot = S_ * B_;   // 8192
constexpr int Ntot = OUT_;      // 4096
constexpr int Ktot = H_;        // 1024

constexpr int BM = 128;         // block tile M
constexpr int BN = 128;         // block tile N
constexpr int BK = 32;          // K tile == WMMA K
constexpr int LDT = BK + 8;     // padded LDS row stride (bf16 elems) -> 80B rows, bank-spread
}

union Frag {
  v16bf v;
  v8bf  h[2];
};

__device__ __forceinline__ v8f wmma_bf16(v16bf a, v16bf b, v8f c) {
  // 8 args: (neg_a, A, neg_b, B, c_mod, C, reuse_a, reuse_b)
  return __builtin_amdgcn_wmma_f32_16x16x32_bf16(false, a, false, b, (short)0, c,
                                                 false, false);
}

__global__ __launch_bounds__(256)
void cpl_gemm_bf16x3(const float* __restrict__ A,   // [M, K]
                     const float* __restrict__ W,   // [N, K]
                     float* __restrict__ C)         // [M, N]
{
  __shared__ bf16_t Ahi[BM * LDT];
  __shared__ bf16_t Alo[BM * LDT];
  __shared__ bf16_t Bhi[BN * LDT];
  __shared__ bf16_t Blo[BN * LDT];

  const int tid  = threadIdx.x;
  const int lane = tid & 31;
  const int wave = tid >> 5;     // 8 waves
  const int wm   = wave & 1;     // 2 wave-rows  -> 64 M each
  const int wn   = wave >> 1;    // 4 wave-cols  -> 32 N each
  const int m0   = blockIdx.y * BM;
  const int n0   = blockIdx.x * BN;

  const int fr  = lane & 15;           // fragment row (M for A, N for B)
  const int akb = (lane >> 4) << 3;    // A K-base per lane group: 0 or 8
  const int bkb = (lane >> 4) << 3;    // B K-base per lane group: 0 or 8

  v8f acc[4][2];
  const v8f vzero = {0.f, 0.f, 0.f, 0.f, 0.f, 0.f, 0.f, 0.f};
#pragma unroll
  for (int i = 0; i < 4; ++i)
#pragma unroll
    for (int j = 0; j < 2; ++j)
      acc[i][j] = vzero;

  for (int k0 = 0; k0 < Ktot; k0 += BK) {
    // ---- Stage A/W tiles: global fp32 -> (hi,lo) bf16 in LDS --------------
    // Tile = 128 rows x 32 fp32. 256 threads x 4 float4 each per matrix.
#pragma unroll
    for (int i = 0; i < 4; ++i) {
      const int g   = tid + i * 256;   // 0..1023
      const int row = g >> 3;          // 0..127
      const int ks  = (g & 7) << 2;    // 0,4,...,28
      {
        const float4 a = *(const float4*)(A + (size_t)(m0 + row) * Ktot + (k0 + ks));
        v4bf hv, lv;
        float f;
        f = a.x; hv[0] = (__bf16)f; lv[0] = (__bf16)(f - (float)hv[0]);
        f = a.y; hv[1] = (__bf16)f; lv[1] = (__bf16)(f - (float)hv[1]);
        f = a.z; hv[2] = (__bf16)f; lv[2] = (__bf16)(f - (float)hv[2]);
        f = a.w; hv[3] = (__bf16)f; lv[3] = (__bf16)(f - (float)hv[3]);
        *(v4bf*)(&Ahi[row * LDT + ks]) = hv;
        *(v4bf*)(&Alo[row * LDT + ks]) = lv;
      }
      {
        const float4 b = *(const float4*)(W + (size_t)(n0 + row) * Ktot + (k0 + ks));
        v4bf hv, lv;
        float f;
        f = b.x; hv[0] = (__bf16)f; lv[0] = (__bf16)(f - (float)hv[0]);
        f = b.y; hv[1] = (__bf16)f; lv[1] = (__bf16)(f - (float)hv[1]);
        f = b.z; hv[2] = (__bf16)f; lv[2] = (__bf16)(f - (float)hv[2]);
        f = b.w; hv[3] = (__bf16)f; lv[3] = (__bf16)(f - (float)hv[3]);
        *(v4bf*)(&Bhi[row * LDT + ks]) = hv;
        *(v4bf*)(&Blo[row * LDT + ks]) = lv;
      }
    }

    // Hint the next K-tile into cache while we compute this one.
    if (k0 + BK < Ktot) {
      const int row = tid >> 3;
      const int ks  = (tid & 7) << 2;
      __builtin_prefetch(A + (size_t)(m0 + row) * Ktot + (k0 + BK + ks), 0, 3);
      __builtin_prefetch(W + (size_t)(n0 + row) * Ktot + (k0 + BK + ks), 0, 3);
    }

    __syncthreads();

    // ---- Load WMMA fragments from LDS -------------------------------------
    // A (16x32 bf16): lane = M row (0..15 twice); lanes 0-15 hold K 0-7 & 16-23,
    // lanes 16-31 hold K 8-15 & 24-31 (two 16B chunks per lane).
    Frag ah[4], al[4], bh[2], bl[2];
#pragma unroll
    for (int i = 0; i < 4; ++i) {
      const int r = (wm * 64 + i * 16 + fr) * LDT;
      ah[i].h[0] = *(const v8bf*)(&Ahi[r + akb]);
      ah[i].h[1] = *(const v8bf*)(&Ahi[r + akb + 16]);
      al[i].h[0] = *(const v8bf*)(&Alo[r + akb]);
      al[i].h[1] = *(const v8bf*)(&Alo[r + akb + 16]);
    }
    // B (32x16 bf16): lane = N col; lanes 0-15 hold K 0-7 & 16-23, lanes 16-31
    // hold K 8-15 & 24-31 (two 16B chunks per lane).
#pragma unroll
    for (int j = 0; j < 2; ++j) {
      const int r = (wn * 32 + j * 16 + fr) * LDT;
      bh[j].h[0] = *(const v8bf*)(&Bhi[r + bkb]);
      bh[j].h[1] = *(const v8bf*)(&Bhi[r + bkb + 16]);
      bl[j].h[0] = *(const v8bf*)(&Blo[r + bkb]);
      bl[j].h[1] = *(const v8bf*)(&Blo[r + bkb + 16]);
    }

    // ---- bf16x3 accumulate: acc += Al*Bh + Ah*Bl + Ah*Bh ------------------
#pragma unroll
    for (int i = 0; i < 4; ++i)
#pragma unroll
      for (int j = 0; j < 2; ++j) {
        v8f c = acc[i][j];
        c = wmma_bf16(al[i].v, bh[j].v, c);
        c = wmma_bf16(ah[i].v, bl[j].v, c);
        c = wmma_bf16(ah[i].v, bh[j].v, c);
        acc[i][j] = c;
      }

    __syncthreads();
  }

  // ---- Epilogue: f32 C/D layout -> global ---------------------------------
  // VGPR r: lanes 0-15 -> (M=r,       N=lane),
  //         lanes 16-31-> (M=r+8,     N=lane-16)
  const int col   = lane & 15;
  const int rbase = (lane >> 4) << 3;
#pragma unroll
  for (int i = 0; i < 4; ++i)
#pragma unroll
    for (int j = 0; j < 2; ++j) {
      const int m = m0 + wm * 64 + i * 16 + rbase;
      const int n = n0 + wn * 32 + j * 16 + col;
      float* p = C + (size_t)m * Ntot + n;
#pragma unroll
      for (int r = 0; r < 8; ++r)
        p[(size_t)r * Ntot] = acc[i][j][r];
    }
}

extern "C" void kernel_launch(void* const* d_in, const int* in_sizes, int n_in,
                              void* d_out, int out_size, void* d_ws, size_t ws_size,
                              hipStream_t stream) {
  const float* input_ = (const float*)d_in[0];   // [S, B, H] == [M, K]
  const float* weight = (const float*)d_in[1];   // [OUT, H]  == [N, K]
  float* out = (float*)d_out;                    // [S, B, OUT] == [M, N]

  dim3 grid(Ntot / BN, Mtot / BM);   // (32, 64)
  dim3 block(256);
  hipLaunchKernelGGL(cpl_gemm_bf16x3, grid, block, 0, stream, input_, weight, out);
}
